// SelfAttention_35596688949359
// MI455X (gfx1250) — matrix operand order repaired and checked by reading, not compiled, not run on hardware
//
#include <hip/hip_runtime.h>
#include <hip/hip_bf16.h>
#include <math.h>

typedef __attribute__((ext_vector_type(16))) _Float16 v16h;
typedef __attribute__((ext_vector_type(8)))  _Float16 v8h;
typedef __attribute__((ext_vector_type(8)))  float    v8f;
typedef __attribute__((ext_vector_type(4)))  unsigned int v4u;
typedef __attribute__((ext_vector_type(8)))  int      v8i;
typedef __attribute__((ext_vector_type(4)))  int      v4i;

#if defined(__gfx1250__) && __has_builtin(__builtin_amdgcn_tensor_load_to_lds)
#define USE_TDM 1
#else
#define USE_TDM 0
#endif

// ---------------------------------------------------------------------------
// f32 -> f16 conversion (grid-stride)
// ---------------------------------------------------------------------------
__global__ __launch_bounds__(256) void f32_to_f16(const float* __restrict__ in,
                                                  _Float16* __restrict__ out,
                                                  long n) {
    long i = (long)blockIdx.x * blockDim.x + threadIdx.x;
    long stride = (long)gridDim.x * blockDim.x;
    for (; i < n; i += stride) out[i] = (_Float16)in[i];
}

#if USE_TDM
// ---------------------------------------------------------------------------
// Issue one TDM 2D tile load: rows x 32 f16 elements, row stride `ld` elems,
// into LDS at byte offset lds_byte_off with a 16-DWORD-interval / 8-DWORD pad
// (=> LDS row stride of 48 halfs / 96 bytes, matching the fragment loaders).
// D# packing per CDNA5 ISA 8.3/8.4 (group0 128b, group1 256b).
// ---------------------------------------------------------------------------
__device__ __forceinline__ void tdm_tile_load(unsigned lds_byte_off,
                                              const _Float16* gptr,
                                              unsigned rows, long ld,
                                              long tensor_cols) {
    unsigned long long ga = (unsigned long long)(uintptr_t)gptr;
    v4u g0;
    g0[0] = 1u;                                            // count=1, user D#
    g0[1] = lds_byte_off;                                  // lds_addr
    g0[2] = (unsigned)(ga & 0xffffffffu);                  // global_addr[31:0]
    g0[3] = (unsigned)((ga >> 32) & 0x01ffffffu)           // global_addr[56:32]
          | (2u << 30);                                    // type = 2 ("image")

    unsigned td0 = (unsigned)tensor_cols;                  // tensor_dim0
    unsigned td1 = 0x00100000u;                            // tensor_dim1 (large)
    unsigned long long st0 = (unsigned long long)ld;       // tensor_dim0_stride

    v8i g1;
    g1[0] = (int)((1u << 16)         // data_size = 1 (2 bytes)
                | (1u << 20)         // pad_enable
                | (3u << 22)         // pad_interval: 16 DWORDs (one 64B row)
                | (7u << 25));       // pad_amount: 8 DWORDs (32B) -> 96B stride
    g1[1] = (int)((td0 & 0xffffu) << 16);                            // dim0 lo
    g1[2] = (int)(((td0 >> 16) & 0xffffu) | ((td1 & 0xffffu) << 16));
    g1[3] = (int)(((td1 >> 16) & 0xffffu) | (32u << 16));            // tile_dim0=32
    g1[4] = (int)(rows & 0xffffu);                                   // tile_dim1
    g1[5] = (int)(st0 & 0xffffffffu);
    g1[6] = (int)((st0 >> 32) & 0xffffu);
    g1[7] = 0;

    v4i gz = {0, 0, 0, 0};
#if __clang_major__ >= 23
    v8i gz8 = {0, 0, 0, 0, 0, 0, 0, 0};
    __builtin_amdgcn_tensor_load_to_lds(g0, g1, gz, gz, gz8, 0);
#else
    __builtin_amdgcn_tensor_load_to_lds(g0, g1, gz, gz, 0);
#endif
}
#endif  // USE_TDM

// ---------------------------------------------------------------------------
// WMMA GEMM.
//   NT==0 : C[m][n] = sum_k A[m][k] * B[k][n]
//   NT==1 : C[m][n] = sum_k A[m][k] * B[n][k]      (for Q @ K^T)
//   OUT_F16 : store _Float16, else float
//   BIAS    : add bias[n] (f32)
// Block = 256 threads = 8 waves. Block tile 128(M) x 128(N), K step 32.
// Wave grid 4x2; each wave computes 32x64 via 2x4 v_wmma_f32_16x16x32_f16.
// Requires M%128==0, N%128==0, K%32==0 (true for all launches below).
// blockIdx.z selects batch via sA/sB/sC element strides.
// A tile (and B tile in NT mode) staged by the Tensor Data Mover when
// available; otherwise by vectorized b128 loads.
// ---------------------------------------------------------------------------
template <int NT, int OUT_F16, int BIAS>
__global__ __launch_bounds__(256) void gemm_wmma(
    const _Float16* __restrict__ A, const _Float16* __restrict__ B,
    const float* __restrict__ bias, void* __restrict__ Cout,
    int M, int N, int K,
    long lda, long ldb, long ldc,
    long sA, long sB, long sC,
    float scale)
{
    // LDS tiles. Row stride 48 halfs (96B): multiple of 16B for b128 access.
    __shared__ _Float16 As[128 * 48];
    __shared__ _Float16 Bs[128 * 48];   // stored N-major: Bs[n][k]

    const int tid  = threadIdx.x;
    const int lane = tid & 31;
    const int wave = tid >> 5;
    const int wm   = wave >> 1;        // 0..3  -> 32-row slab
    const int wn   = wave & 1;         // 0..1  -> 64-col slab
    const int gm0  = blockIdx.y * 128;
    const int gn0  = blockIdx.x * 128;
    const int z    = blockIdx.z;

    const _Float16* Ab = A + (size_t)z * sA;
    const _Float16* Bb = B + (size_t)z * sB;

    const int half = lane >> 4;        // lane group 0..15 / 16..31
    const int r15  = lane & 15;

    v8f zero = {};
    v8f acc[2][4];
#pragma unroll
    for (int i = 0; i < 2; ++i)
#pragma unroll
        for (int j = 0; j < 4; ++j) acc[i][j] = zero;

    union V16 { v16h v; v8h h[2]; };

    for (int k0 = 0; k0 < K; k0 += 32) {
        __syncthreads();   // protect LDS from previous iteration's readers

#if USE_TDM
        // ---- stage via Tensor Data Mover (wave 0 issues the DMA) ----
        if (wave == 0) {
            tdm_tile_load((unsigned)(uintptr_t)As,
                          Ab + (size_t)gm0 * lda + k0, 128, lda, (long)K);
            if (NT)
                tdm_tile_load((unsigned)(uintptr_t)Bs,
                              Bb + (size_t)gn0 * ldb + k0, 128, ldb, (long)K);
        }
#else
        // ---- stage A tile: 128 x 32, row-major, b128 global loads ----
        {
            int r  = tid >> 1;
            int cc = (tid & 1) * 16;
            const _Float16* g = Ab + (size_t)(gm0 + r) * lda + k0 + cc;
            *reinterpret_cast<v8h*>(&As[r * 48 + cc])     = *reinterpret_cast<const v8h*>(g);
            *reinterpret_cast<v8h*>(&As[r * 48 + cc + 8]) = *reinterpret_cast<const v8h*>(g + 8);
        }
        if (NT) {
            // B is [N x K] row-major: rows copy straight in (no transpose)
            int r  = tid >> 1;
            int cc = (tid & 1) * 16;
            const _Float16* g = Bb + (size_t)(gn0 + r) * ldb + k0 + cc;
            *reinterpret_cast<v8h*>(&Bs[r * 48 + cc])     = *reinterpret_cast<const v8h*>(g);
            *reinterpret_cast<v8h*>(&Bs[r * 48 + cc + 8]) = *reinterpret_cast<const v8h*>(g + 8);
        }
#endif
        if (!NT) {
            // B is [K x N] row-major: transpose on LDS store
            int kk  = tid & 31;
            int n16 = (tid >> 5) * 16;
            const _Float16* g = Bb + (size_t)(k0 + kk) * ldb + gn0 + n16;
            v8h blo = *reinterpret_cast<const v8h*>(g);
            v8h bhi = *reinterpret_cast<const v8h*>(g + 8);
#pragma unroll
            for (int i = 0; i < 8; ++i) Bs[(n16 + i)     * 48 + kk] = blo[i];
#pragma unroll
            for (int i = 0; i < 8; ++i) Bs[(n16 + 8 + i) * 48 + kk] = bhi[i];
        }
#if USE_TDM
        if (wave == 0) __builtin_amdgcn_s_wait_tensorcnt(0);
#endif
        __syncthreads();

        // ---- fragment loads (documented gfx1250 wave32 layouts) ----
        V16 af[2], bf[4];
#pragma unroll
        for (int i = 0; i < 2; ++i) {
            int m = wm * 32 + i * 16 + r15;
            // lanes 0-15: K 0..7 and 16..23 ; lanes 16-31: K 8..15 and 24..31
            af[i].h[0] = *reinterpret_cast<const v8h*>(&As[m * 48 + 8 * half]);
            af[i].h[1] = *reinterpret_cast<const v8h*>(&As[m * 48 + 16 + 8 * half]);
        }
#pragma unroll
        for (int j = 0; j < 4; ++j) {
            int n = wn * 64 + j * 16 + r15;
            // lanes 0-15: K 0..7 and 16..23 ; lanes 16-31: K 8..15 and 24..31
            bf[j].h[0] = *reinterpret_cast<const v8h*>(&Bs[n * 48 + 8 * half]);
            bf[j].h[1] = *reinterpret_cast<const v8h*>(&Bs[n * 48 + 16 + 8 * half]);
        }

#pragma unroll
        for (int i = 0; i < 2; ++i)
#pragma unroll
            for (int j = 0; j < 4; ++j)
                acc[i][j] = __builtin_amdgcn_wmma_f32_16x16x32_f16(
                    false, af[i].v, false, bf[j].v,
                    (short)0, acc[i][j], false, false);
    }

    // ---- store: C/D layout m = vgpr + 8*half, n = lane&15 ----
#pragma unroll
    for (int i = 0; i < 2; ++i)
#pragma unroll
        for (int j = 0; j < 4; ++j) {
            int nglob = gn0 + wn * 64 + j * 16 + r15;
            float badd = BIAS ? bias[nglob] : 0.0f;
#pragma unroll
            for (int v = 0; v < 8; ++v) {
                int m = gm0 + wm * 32 + i * 16 + v + 8 * half;
                float val = acc[i][j][v] * scale + badd;
                size_t idx = (size_t)z * sC + (size_t)m * ldc + nglob;
                if (OUT_F16)
                    reinterpret_cast<_Float16*>(Cout)[idx] = (_Float16)val;
                else
                    reinterpret_cast<float*>(Cout)[idx] = val;
            }
        }
}

// ---------------------------------------------------------------------------
// Row softmax, in place on f16 scores. One block per row of length N.
// ---------------------------------------------------------------------------
__global__ __launch_bounds__(256) void softmax_rows(_Float16* __restrict__ S, int N) {
    __shared__ float red[256];
    const int t = threadIdx.x;
    _Float16* p = S + (size_t)blockIdx.x * N;

    float mx = -INFINITY;
    for (int i = t; i < N; i += 256) mx = fmaxf(mx, (float)p[i]);
    red[t] = mx;
    __syncthreads();
    for (int s = 128; s > 0; s >>= 1) {
        if (t < s) red[t] = fmaxf(red[t], red[t + s]);
        __syncthreads();
    }
    mx = red[0];
    __syncthreads();

    float sum = 0.0f;
    for (int i = t; i < N; i += 256) {
        float e = __expf((float)p[i] - mx);
        sum += e;
        p[i] = (_Float16)e;
    }
    red[t] = sum;
    __syncthreads();
    for (int s = 128; s > 0; s >>= 1) {
        if (t < s) red[t] += red[t + s];
        __syncthreads();
    }
    float inv = 1.0f / red[0];
    for (int i = t; i < N; i += 256) p[i] = (_Float16)((float)p[i] * inv);
}

// ---------------------------------------------------------------------------
// Launch: X[4,2048,1024], W_in[1024,3072], b_in[3072], W_out[1024,1024], b_out[1024]
// ---------------------------------------------------------------------------
extern "C" void kernel_launch(void* const* d_in, const int* in_sizes, int n_in,
                              void* d_out, int out_size, void* d_ws, size_t ws_size,
                              hipStream_t stream) {
    (void)in_sizes; (void)n_in; (void)out_size; (void)ws_size;

    const float* X     = (const float*)d_in[0];
    const float* W_in  = (const float*)d_in[1];
    const float* b_in  = (const float*)d_in[2];
    const float* W_out = (const float*)d_in[3];
    const float* b_out = (const float*)d_in[4];

    const int  Bz = 4, S = 2048, D = 1024;
    const long MS = (long)Bz * S;          // 8192 total rows

    char* ws = (char*)d_ws;
    const size_t MB = 1024ull * 1024ull;
    _Float16* Xh    = (_Float16*)(ws);                 // 8192*1024        (16 MB)
    _Float16* Winh  = (_Float16*)(ws + 16 * MB);       // 1024*3072        ( 6 MB)
    _Float16* Wouth = (_Float16*)(ws + 22 * MB);       // 1024*1024        ( 2 MB)
    _Float16* QKV   = (_Float16*)(ws + 24 * MB);       // 8192*3072        (48 MB)
    _Float16* Sc    = (_Float16*)(ws + 72 * MB);       // 4*2048*2048      (32 MB)
    _Float16* Attn  = (_Float16*)(ws + 104 * MB);      // 8192*1024        (16 MB)

    // 1) precision conversion
    f32_to_f16<<<4096, 256, 0, stream>>>(X,     Xh,    MS * D);
    f32_to_f16<<<2048, 256, 0, stream>>>(W_in,  Winh,  (long)D * 3 * D);
    f32_to_f16<<<1024, 256, 0, stream>>>(W_out, Wouth, (long)D * D);

    // 2) QKV = Xh @ Winh + b_in        [8192 x 3072], f16 out
    gemm_wmma<0, 1, 1><<<dim3(3 * D / 128, MS / 128, 1), 256, 0, stream>>>(
        Xh, Winh, b_in, QKV,
        (int)MS, 3 * D, D,
        D, 3L * D, 3L * D,
        0, 0, 0, 1.0f);

    // 3) scores = (Q @ K^T) / sqrt(D)  per batch   [2048 x 2048], f16 out
    gemm_wmma<1, 1, 0><<<dim3(S / 128, S / 128, Bz), 256, 0, stream>>>(
        QKV, QKV + D, nullptr, Sc,
        S, S, D,
        3L * D, 3L * D, (long)S,
        (long)S * 3 * D, (long)S * 3 * D, (long)S * S,
        0.03125f /* 1/sqrt(1024) */);

    // 4) softmax rows, in place on Sc
    softmax_rows<<<Bz * S, 256, 0, stream>>>(Sc, S);

    // 5) attn = P @ V   per batch     [2048 x 1024], f16 out
    gemm_wmma<0, 1, 0><<<dim3(D / 128, S / 128, Bz), 256, 0, stream>>>(
        Sc, QKV + 2 * D, nullptr, Attn,
        S, D, S,
        (long)S, 3L * D, (long)D,
        (long)S * S, (long)S * 3 * D, (long)S * D,
        1.0f);

    // 6) out = attn @ W_out + b_out   [8192 x 1024], f32 out -> d_out
    gemm_wmma<0, 0, 1><<<dim3(D / 128, MS / 128, 1), 256, 0, stream>>>(
        Attn, Wouth, b_out, d_out,
        (int)MS, D, D,
        (long)D, (long)D, (long)D,
        0, 0, 0, 1.0f);
}
